// AnyNet_8323646619835
// MI455X (gfx1250) — matrix operand order repaired and checked by reading, not compiled, not run on hardware
//
#include <hip/hip_runtime.h>
#include <hip/hip_bf16.h>
#include <math.h>

typedef __attribute__((ext_vector_type(16))) __bf16 v16bf;
typedef __attribute__((ext_vector_type(8)))  __bf16 v8bf;
typedef __attribute__((ext_vector_type(8)))  float  v8f;
typedef __attribute__((ext_vector_type(4)))  int    v4i;
typedef __attribute__((ext_vector_type(8)))  int    v8i;

// ---------------- problem constants (from reference) ----------------
constexpr int BN = 8;                 // batch
constexpr int FC = 8;                 // feature channels
constexpr int D0 = 12, H0 = 32,  W0 = 80;    // scale 0
constexpr int S1 = 5,  H1 = 64,  W1 = 160;   // scale 1 (2*3-1 shifts)
constexpr int S2 = 5,  H2 = 128, W2 = 320;   // scale 2
constexpr int IMGH = 512, IMGW = 1280;
constexpr float BASE_FOCAL = 0.2f * 320.0f;  // 64
constexpr float MIND = 0.1f, MAXD = 100.0f;

// =====================================================================
// Scale-0 L1 cost volume: cost[b,d,h,w] = sum_c |fl - fr(w-d)|, zero pad
// =====================================================================
__global__ void cost_l1_kernel(const float* __restrict__ fl,
                               const float* __restrict__ fr,
                               float* __restrict__ cost) {
  int idx = blockIdx.x * blockDim.x + threadIdx.x;
  const int total = BN * D0 * H0 * W0;
  if (idx >= total) return;
  int w = idx % W0;
  int h = (idx / W0) % H0;
  int d = (idx / (W0 * H0)) % D0;
  int b = idx / (W0 * H0 * D0);
  int sw = w - d;
  float s = 0.f;
  for (int c = 0; c < FC; ++c) {
    float a = fl[((b * FC + c) * H0 + h) * W0 + w];
    float r = (sw >= 0) ? fr[((b * FC + c) * H0 + h) * W0 + sw] : 0.f;
    s += fabsf(a - r);
  }
  cost[idx] = s;
}

// =====================================================================
// Residual cost volume (scales 1,2): bilinear warp of fr by disp-shift
// =====================================================================
__global__ void cost_resid_kernel(const float* __restrict__ fl,
                                  const float* __restrict__ fr,
                                  const float* __restrict__ disp,
                                  float* __restrict__ cost,
                                  int H, int W, int S, float shift0) {
  int idx = blockIdx.x * blockDim.x + threadIdx.x;
  const int total = BN * S * H * W;
  if (idx >= total) return;
  int w = idx % W;
  int h = (idx / W) % H;
  int s = (idx / (W * H)) % S;
  int b = idx / (W * H * S);
  float shift = shift0 + (float)s;
  float sd = disp[(b * H + h) * W + w] - shift;
  float xs = (float)w - sd;
  float x0f = floorf(xs);
  float w1 = xs - x0f;
  int x0 = (int)x0f;
  int x1 = x0 + 1;
  float m0 = (x0 >= 0 && x0 <= W - 1) ? 1.f : 0.f;
  float m1 = (x1 >= 0 && x1 <= W - 1) ? 1.f : 0.f;
  int x0c = min(max(x0, 0), W - 1);
  int x1c = min(max(x1, 0), W - 1);
  float acc = 0.f;
  for (int c = 0; c < FC; ++c) {
    const float* row = fr + ((b * FC + c) * H + h) * W;
    float v = row[x0c] * m0 * (1.f - w1) + row[x1c] * m1 * w1;
    acc += fabsf(fl[((b * FC + c) * H + h) * W + w] - v);
  }
  cost[idx] = acc;
}

// =====================================================================
// Scale-0 input conv 1->16, 3x3x3 SAME, ReLU.  One WMMA per 16-wide tile:
// M = 16 output channels, N = 16 w-positions, K = 27 taps padded to 32.
// =====================================================================
__global__ void conv_in16_wmma_kernel(const float* __restrict__ cost,  // [B][D0][H0][W0]
                                      const float* __restrict__ wgt,   // [16][27]
                                      float* __restrict__ y) {         // [B][16][D0][H0][W0]
  const int lane = threadIdx.x & 31;
  const int warp = threadIdx.x >> 5;
  const int lo = lane & 15, hi = lane >> 4;
  int tile = blockIdx.x * (blockDim.x >> 5) + warp;   // 15360 tiles exactly
  int wt = tile % (W0 / 16);
  int h  = (tile / (W0 / 16)) % H0;
  int d  = (tile / ((W0 / 16) * H0)) % D0;
  int b  = tile / ((W0 / 16) * H0 * D0);
  int w0 = wt * 16;

  v16bf a, bm;
  #pragma unroll
  for (int e = 0; e < 8; ++e) {
    int k0 = hi * 8 + e;                         // < 16, always a valid tap
    a[e] = (__bf16)wgt[lo * 27 + k0];
    int k1 = 16 + hi * 8 + e;                    // 16..31
    a[e + 8] = (k1 < 27) ? (__bf16)wgt[lo * 27 + k1] : (__bf16)0.f;
  }
  #pragma unroll
  for (int e = 0; e < 16; ++e) {
    int k = (e < 8) ? hi * 8 + e : 16 + hi * 8 + (e - 8);  // tap index for this lane half
    int kk = (k < 27) ? k : 0;
    int kd = kk / 9, kh = (kk / 3) % 3, kw = kk % 3;
    int gd = d + kd - 1, gh = h + kh - 1, gw = w0 + lo + kw - 1;
    bool inb = (k < 27) && (gd >= 0) && (gd < D0) && (gh >= 0) && (gh < H0) &&
               (gw >= 0) && (gw < W0);
    float v = inb ? cost[((b * D0 + gd) * H0 + gh) * W0 + gw] : 0.f;
    bm[e] = (__bf16)v;
  }
  v8f acc = {0.f, 0.f, 0.f, 0.f, 0.f, 0.f, 0.f, 0.f};
  acc = __builtin_amdgcn_wmma_f32_16x16x32_bf16(false, a, false, bm,
                                                (short)0, acc, false, false);
  #pragma unroll
  for (int r = 0; r < 8; ++r) {
    int co = hi * 8 + r;
    y[(((b * 16 + co) * D0 + d) * H0 + h) * W0 + w0 + lo] = fmaxf(acc[r], 0.f);
  }
}

// =====================================================================
// Scale-0 mid conv 16->16, 3x3x3 SAME, ReLU.  Implicit GEMM:
// 14 WMMAs per tile (tap pairs, K=32 = 2 taps x 16 Cin).
//  - f32 weights DMA'd into LDS once per block by the Tensor Data Mover
//    (tensor_load_to_lds + s_wait_tensorcnt), then converted to bf16.
//  - Activation halo staged per wave as bf16 with a [dd][hh][ww][ci]
//    layout so A and B fragments are built with ds_load_b128.
// =====================================================================
constexpr int WCNT = 16 * 16 * 27;               // 6912 f32 weights

__global__ void conv_mid16_wmma_kernel(const float* __restrict__ x,    // [B][16][D0][H0][W0]
                                       const float* __restrict__ wgt,  // [16][16][27]
                                       float* __restrict__ y) {
  __shared__ alignas(16) float  wf[WCNT];                 // raw f32 weights (TDM dest)
  __shared__ alignas(32) __bf16 wlds[16 * 28 * 16];       // [co][tap(28, slot27=0)][ci]
  __shared__ alignas(32) __bf16 xlds[4][3 * 3 * 18 * 16]; // [dd][hh][ww(18)][ci]
  const int tid = threadIdx.x;
  const int lane = tid & 31, warp = tid >> 5;
  const int lo = lane & 15, hi = lane >> 4;

  // ---- TDM: DMA the whole weight block (flat 1-D tile) into LDS ----
  if (warp == 0) {
    unsigned lds_off = (unsigned)(size_t)(void*)&wf[0];
    unsigned long long ga = (unsigned long long)(size_t)wgt;
    v4i g0;
    g0[0] = 1;                                            // count=1 (valid user D#)
    g0[1] = (int)lds_off;                                 // lds_addr (bytes)
    g0[2] = (int)(ga & 0xffffffffu);                      // global_addr[31:0]
    g0[3] = (int)(((ga >> 32) & 0x01ffffffu) | (2u << 30)); // addr[56:32] | type=2
    v8i g1;
    g1[0] = 0x00020000;                                   // data_size=4B, no multicast
    g1[1] = (int)((unsigned)WCNT << 16);                  // tensor_dim0[15:0]
    g1[2] = 0x00010000;                                   // tensor_dim0 hi=0, tensor_dim1=1
    g1[3] = (int)((unsigned)WCNT << 16);                  // tile_dim0 = 6912
    g1[4] = 0;                                            // tile_dim1/2 unused
    g1[5] = WCNT;                                         // tensor_dim0_stride
    g1[6] = 0;
    g1[7] = 0;
    asm volatile("tensor_load_to_lds %0, %1"
                 :: "s"(g0), "s"(g1) : "memory");
    __builtin_amdgcn_s_wait_tensorcnt(0);
  }
  __syncthreads();                                        // wf valid for all waves

  // ---- convert weights f32 -> bf16, [co][tap pad 28][ci], slot 27 = 0 ----
  for (int i = tid; i < 16 * 28 * 16; i += 128) {
    int ci = i & 15;
    int t  = (i >> 4) % 28;
    int co = i / (28 * 16);
    wlds[i] = (t < 27) ? (__bf16)wf[(co * 16 + ci) * 27 + t] : (__bf16)0.f;
  }

  int tile = blockIdx.x * 4 + warp;              // 15360 tiles exactly
  int wt = tile % (W0 / 16);
  int h  = (tile / (W0 / 16)) % H0;
  int d  = (tile / ((W0 / 16) * H0)) % D0;
  int b  = tile / ((W0 / 16) * H0 * D0);
  int w0 = wt * 16;

  // ---- stage input halo [3][3][18] x 16ci, ww fastest (coalesced loads),
  //      LDS store transposed so ci is contiguous (32B per B element) ----
  for (int i = lane; i < 16 * 3 * 3 * 18; i += 32) {
    int ww = i % 18;
    int hh = (i / 18) % 3;
    int dd = (i / 54) % 3;
    int ci = i / 162;
    int gd = d + dd - 1, gh = h + hh - 1, gw = w0 + ww - 1;
    bool inb = (gd >= 0) && (gd < D0) && (gh >= 0) && (gh < H0) &&
               (gw >= 0) && (gw < W0);
    float v = inb ? x[(((b * 16 + ci) * D0 + gd) * H0 + gh) * W0 + gw] : 0.f;
    xlds[warp][((dd * 3 + hh) * 18 + ww) * 16 + ci] = (__bf16)v;
  }
  __syncthreads();

  v8f acc = {0.f, 0.f, 0.f, 0.f, 0.f, 0.f, 0.f, 0.f};
  #pragma unroll
  for (int p = 0; p < 14; ++p) {
    const int t0 = 2 * p, t1 = 2 * p + 1;        // t1==27 on last pair -> padded
    // A fragment: one aligned 32B chunk (ci 0..15 of tap t0 for lanes 0-15, t1 for 16-31)
    v16bf a = *(const v16bf*)&wlds[(lo * 28 + (hi ? t1 : t0)) * 16];  // slot 27 zeroed
    // B fragment: one aligned 32B chunk of 16 ci values
    int t = hi ? t1 : t0;                        // lanes 0-15: even tap, 16-31: odd
    int tt = (t < 27) ? t : 0;
    int kd = tt / 9, kh = (tt / 3) % 3, kw = tt % 3;
    v16bf bm = *(const v16bf*)&xlds[warp][((kd * 3 + kh) * 18 + lo + kw) * 16];
    #pragma unroll
    for (int e = 0; e < 16; ++e)
      bm[e] = (t < 27) ? bm[e] : (__bf16)0.f;    // zero the pad tap (hi half, p=13)
    acc = __builtin_amdgcn_wmma_f32_16x16x32_bf16(false, a, false, bm,
                                                  (short)0, acc, false, false);
  }
  #pragma unroll
  for (int r = 0; r < 8; ++r) {
    int co = hi * 8 + r;
    y[(((b * 16 + co) * D0 + d) * H0 + h) * W0 + w0 + lo] = fmaxf(acc[r], 0.f);
  }
}

// =====================================================================
// Generic direct 3D conv (used where M would waste the matrix pipe:
// 4-channel scales and the 16->1 output conv).
// =====================================================================
template <int CI, int CO, bool RELU>
__global__ void conv3d_direct_kernel(const float* __restrict__ x,
                                     const float* __restrict__ w,
                                     float* __restrict__ y,
                                     int D, int H, int W) {
  int idx = blockIdx.x * blockDim.x + threadIdx.x;
  int total = BN * CO * D * H * W;
  if (idx >= total) return;
  int ww = idx % W;
  int h  = (idx / W) % H;
  int d  = (idx / (W * H)) % D;
  int co = (idx / (W * H * D)) % CO;
  int b  = idx / (W * H * D * CO);
  float acc = 0.f;
  for (int ci = 0; ci < CI; ++ci) {
    const float* xp = x + ((size_t)(b * CI + ci) * D) * H * W;
    const float* wp = w + (co * CI + ci) * 27;
    for (int kd = 0; kd < 3; ++kd) {
      int gd = d + kd - 1;
      if (gd < 0 || gd >= D) continue;
      for (int kh = 0; kh < 3; ++kh) {
        int gh = h + kh - 1;
        if (gh < 0 || gh >= H) continue;
        for (int kw = 0; kw < 3; ++kw) {
          int gw = ww + kw - 1;
          if (gw < 0 || gw >= W) continue;
          acc += xp[(gd * H + gh) * W + gw] * wp[kd * 9 + kh * 3 + kw];
        }
      }
    }
  }
  y[idx] = RELU ? fmaxf(acc, 0.f) : acc;
}

// =====================================================================
// Soft-argmin disparity regression + scale by (img_h / H)
// =====================================================================
__global__ void disp_reg_kernel(const float* __restrict__ cost,  // [B][D][H][W]
                                float* __restrict__ disp,        // [B][H][W]
                                int D, float start, float scale, int H, int W) {
  int idx = blockIdx.x * blockDim.x + threadIdx.x;
  int total = BN * H * W;
  if (idx >= total) return;
  int w = idx % W;
  int h = (idx / W) % H;
  int b = idx / (W * H);
  float c[12];
  float m = -1e30f;
  for (int d = 0; d < D; ++d) {
    c[d] = -cost[((b * D + d) * H + h) * W + w];
    m = fmaxf(m, c[d]);
  }
  float se = 0.f, sw = 0.f;
  for (int d = 0; d < D; ++d) {
    float e = __expf(c[d] - m);
    se += e;
    sw += e * (start + (float)d);
  }
  disp[idx] = scale * (sw / se);
}

// =====================================================================
// Bilinear resize (half-pixel centers, edge clamp), optional scale+add
// =====================================================================
__global__ void resize_bilinear_kernel(const float* __restrict__ src, int hs, int wsz,
                                       float* __restrict__ dst, int hd, int wd,
                                       const float* __restrict__ addend, float mul) {
  int idx = blockIdx.x * blockDim.x + threadIdx.x;
  int total = BN * hd * wd;
  if (idx >= total) return;
  int x = idx % wd;
  int y = (idx / wd) % hd;
  int b = idx / (wd * hd);
  float fy = (y + 0.5f) * (float)hs / (float)hd - 0.5f;
  float fx = (x + 0.5f) * (float)wsz / (float)wd - 0.5f;
  int y0 = (int)floorf(fy); float ty = fy - (float)y0;
  int x0 = (int)floorf(fx); float tx = fx - (float)x0;
  int y0c = min(max(y0, 0), hs - 1), y1c = min(max(y0 + 1, 0), hs - 1);
  int x0c = min(max(x0, 0), wsz - 1), x1c = min(max(x0 + 1, 0), wsz - 1);
  const float* s = src + (size_t)b * hs * wsz;
  float v = (s[y0c * wsz + x0c] * (1.f - tx) + s[y0c * wsz + x1c] * tx) * (1.f - ty) +
            (s[y1c * wsz + x0c] * (1.f - tx) + s[y1c * wsz + x1c] * tx) * ty;
  v *= mul;
  if (addend) v += addend[idx];
  dst[idx] = v;
}

// =====================================================================
// depth = clip(nan_to_num(64 / |pred0|), 0.1, 100)
// =====================================================================
__global__ void depth_kernel(const float* __restrict__ pred0,
                             float* __restrict__ depth) {
  int idx = blockIdx.x * blockDim.x + threadIdx.x;
  const int total = BN * IMGH * IMGW;
  if (idx >= total) return;
  float dsp = fabsf(pred0[idx]);
  float d = BASE_FOCAL / dsp;
  if (isnan(d)) d = MAXD;
  d = fminf(fmaxf(d, MIND), MAXD);
  depth[idx] = d;
}

// =====================================================================
static inline int cdiv(int a, int b) { return (a + b - 1) / b; }

extern "C" void kernel_launch(void* const* d_in, const int* in_sizes, int n_in,
                              void* d_out, int out_size, void* d_ws, size_t ws_size,
                              hipStream_t stream) {
  (void)in_sizes; (void)n_in; (void)out_size; (void)ws_size;
  const float* f_l0  = (const float*)d_in[0];
  const float* f_r0  = (const float*)d_in[1];
  const float* f_l1  = (const float*)d_in[2];
  const float* f_r1  = (const float*)d_in[3];
  const float* f_l2  = (const float*)d_in[4];
  const float* f_r2  = (const float*)d_in[5];
  const float* w_in0  = (const float*)d_in[6];
  const float* w_mid0 = (const float*)d_in[7];
  const float* w_out0 = (const float*)d_in[8];
  const float* w_in1  = (const float*)d_in[9];
  const float* w_mid1 = (const float*)d_in[10];
  const float* w_out1 = (const float*)d_in[11];
  const float* w_in2  = (const float*)d_in[12];
  const float* w_mid2 = (const float*)d_in[13];
  const float* w_out2 = (const float*)d_in[14];

  // ---- workspace layout (floats) ----
  float* ws = (float*)d_ws;
  size_t off = 0;
  const size_t COSTMAX = (size_t)BN * S2 * H2 * W2;     // 1,638,400 (covers all scales)
  const size_t XBUF    = (size_t)BN * 4 * S2 * H2 * W2; // 6,553,600 (>= 8*16*12*32*80)
  float* cost  = ws + off; off += COSTMAX;
  float* costc = ws + off; off += COSTMAX;
  float* xa    = ws + off; off += XBUF;
  float* xb    = ws + off; off += XBUF;
  float* disp  = ws + off; off += (size_t)BN * H2 * W2;
  float* wfl   = ws + off; off += (size_t)BN * H2 * W2;

  const int NPIX = BN * IMGH * IMGW;                    // 5,242,880 per output
  float* pred0 = (float*)d_out;
  float* pred1 = pred0 + NPIX;
  float* pred2 = pred1 + NPIX;
  float* depth = pred2 + NPIX;

  // ================= scale 0 (WMMA path, 16 channels) =================
  cost_l1_kernel<<<cdiv(BN * D0 * H0 * W0, 256), 256, 0, stream>>>(f_l0, f_r0, cost);

  const int TILES0 = BN * D0 * H0 * (W0 / 16);          // 15,360 (4 waves / block)
  conv_in16_wmma_kernel<<<TILES0 / 4, 128, 0, stream>>>(cost, w_in0, xa);
  float *cur = xa, *nxt = xb;
  for (int k = 0; k < 4; ++k) {
    conv_mid16_wmma_kernel<<<TILES0 / 4, 128, 0, stream>>>(
        cur, w_mid0 + (size_t)k * 16 * 16 * 27, nxt);
    float* t = cur; cur = nxt; nxt = t;
  }
  conv3d_direct_kernel<16, 1, false><<<cdiv(BN * D0 * H0 * W0, 256), 256, 0, stream>>>(
      cur, w_out0, costc, D0, H0, W0);
  disp_reg_kernel<<<cdiv(BN * H0 * W0, 256), 256, 0, stream>>>(
      costc, disp, D0, 0.f, (float)IMGH / H0, H0, W0);
  resize_bilinear_kernel<<<cdiv(NPIX, 256), 256, 0, stream>>>(
      disp, H0, W0, pred0, IMGH, IMGW, nullptr, 1.f);

  // ================= scale 1 (direct path, 4 channels) =================
  resize_bilinear_kernel<<<cdiv(BN * H1 * W1, 256), 256, 0, stream>>>(
      pred0, IMGH, IMGW, wfl, H1, W1, nullptr, (float)H1 / IMGH);
  cost_resid_kernel<<<cdiv(BN * S1 * H1 * W1, 256), 256, 0, stream>>>(
      f_l1, f_r1, wfl, cost, H1, W1, S1, -2.f);
  conv3d_direct_kernel<1, 4, true><<<cdiv(BN * 4 * S1 * H1 * W1, 256), 256, 0, stream>>>(
      cost, w_in1, xa, S1, H1, W1);
  cur = xa; nxt = xb;
  for (int k = 0; k < 4; ++k) {
    conv3d_direct_kernel<4, 4, true><<<cdiv(BN * 4 * S1 * H1 * W1, 256), 256, 0, stream>>>(
        cur, w_mid1 + (size_t)k * 4 * 4 * 27, nxt, S1, H1, W1);
    float* t = cur; cur = nxt; nxt = t;
  }
  conv3d_direct_kernel<4, 1, false><<<cdiv(BN * S1 * H1 * W1, 256), 256, 0, stream>>>(
      cur, w_out1, costc, S1, H1, W1);
  disp_reg_kernel<<<cdiv(BN * H1 * W1, 256), 256, 0, stream>>>(
      costc, disp, S1, -2.f, (float)IMGH / H1, H1, W1);
  resize_bilinear_kernel<<<cdiv(NPIX, 256), 256, 0, stream>>>(
      disp, H1, W1, pred1, IMGH, IMGW, pred0, 1.f);

  // ================= scale 2 (direct path, 4 channels) =================
  resize_bilinear_kernel<<<cdiv(BN * H2 * W2, 256), 256, 0, stream>>>(
      pred1, IMGH, IMGW, wfl, H2, W2, nullptr, (float)H2 / IMGH);
  cost_resid_kernel<<<cdiv(BN * S2 * H2 * W2, 256), 256, 0, stream>>>(
      f_l2, f_r2, wfl, cost, H2, W2, S2, -2.f);
  conv3d_direct_kernel<1, 4, true><<<cdiv(BN * 4 * S2 * H2 * W2, 256), 256, 0, stream>>>(
      cost, w_in2, xa, S2, H2, W2);
  cur = xa; nxt = xb;
  for (int k = 0; k < 4; ++k) {
    conv3d_direct_kernel<4, 4, true><<<cdiv(BN * 4 * S2 * H2 * W2, 256), 256, 0, stream>>>(
        cur, w_mid2 + (size_t)k * 4 * 4 * 27, nxt, S2, H2, W2);
    float* t = cur; cur = nxt; nxt = t;
  }
  conv3d_direct_kernel<4, 1, false><<<cdiv(BN * S2 * H2 * W2, 256), 256, 0, stream>>>(
      cur, w_out2, costc, S2, H2, W2);
  disp_reg_kernel<<<cdiv(BN * H2 * W2, 256), 256, 0, stream>>>(
      costc, disp, S2, -2.f, (float)IMGH / H2, H2, W2);
  resize_bilinear_kernel<<<cdiv(NPIX, 256), 256, 0, stream>>>(
      disp, H2, W2, pred2, IMGH, IMGW, pred1, 1.f);

  // ================= depth =================
  depth_kernel<<<cdiv(NPIX, 256), 256, 0, stream>>>(pred0, depth);
}
